// SparseGraphAttentionLayer_40759239639873
// MI455X (gfx1250) — compile-verified, matrix operand order checked by reading, not run on hardware
//
#include <hip/hip_runtime.h>
#include <hip/hip_bf16.h>

typedef __attribute__((ext_vector_type(16))) _Float16 v16h;
typedef __attribute__((ext_vector_type(8)))  float    v8f;

#define D_IN   256
#define D_OUT  64
#define ALPHA  0.2f

// --------------------------------------------------------------------------
// Kernel 1: h = x @ w  via split-f16 WMMA (error-compensated: hi*hi + hi*lo +
// lo*hi with f32 accumulation ~ fp32 accuracy), fused epilogue computes
// s1 = h@a1, s2 = h@a2 with 16-lane shuffle reductions.
//
// Each block = 256 threads = 8 waves. Wave i computes a 16-row x 64-col tile
// (4 WMMA n-tiles, K-loop of 8 steps of 32). w is pre-split into f16 hi/lo
// and staged in LDS pre-swizzled into the CDNA5 16-bit B-operand layout
// (lane = 16*khalf + n, VGPR p holds the f16 pair for k-pair p), so the
// steady state is ds_load_b128 + v_wmma only.
// --------------------------------------------------------------------------
__device__ __forceinline__ int kpair_base(int p, int khalf) {
    // CDNA5 16-bit A/B operand k mapping: VGPR p holds k = {pairs} per half
    return (p < 4) ? (8 * khalf + 2 * p) : (16 + 8 * khalf + 2 * (p - 4));
}

__global__ __launch_bounds__(256)
void gat_gemm_wmma(const float* __restrict__ x, const float* __restrict__ w,
                   const float* __restrict__ a, float* __restrict__ hout,
                   float* __restrict__ s1, float* __restrict__ s2, int n)
{
    // [kstep][ntile][lane][16 f16] : 8*4*32*16 halves = 32 KB each
    __shared__ alignas(32) _Float16 ldsHi[8 * 4 * 32 * 16];
    __shared__ alignas(32) _Float16 ldsLo[8 * 4 * 32 * 16];

    const int tid = threadIdx.x;

    // Cooperative fill: split w into f16 hi/lo, swizzled to B-operand layout.
    for (int idx = tid; idx < 8 * 4 * 32 * 8; idx += 256) {
        const int p     = idx & 7;
        const int lane  = (idx >> 3) & 31;
        const int nt    = (idx >> 8) & 3;
        const int ks    = idx >> 10;
        const int khalf = lane >> 4;
        const int col   = nt * 16 + (lane & 15);
        const int k0    = ks * 32 + kpair_base(p, khalf);
        const float w0  = w[k0 * D_OUT + col];
        const float w1  = w[(k0 + 1) * D_OUT + col];
        const _Float16 h0 = (_Float16)w0, h1 = (_Float16)w1;
        const _Float16 l0 = (_Float16)(w0 - (float)h0);
        const _Float16 l1 = (_Float16)(w1 - (float)h1);
        const int base = ((ks * 4 + nt) * 32 + lane) * 16;
        ldsHi[base + 2 * p]     = h0;
        ldsHi[base + 2 * p + 1] = h1;
        ldsLo[base + 2 * p]     = l0;
        ldsLo[base + 2 * p + 1] = l1;
    }
    __syncthreads();

    const int wave  = tid >> 5;
    const int lane  = tid & 31;
    const int mBase = (blockIdx.x * 8 + wave) * 16;
    if (mBase >= n) return;               // wave-uniform: EXEC all-1s for WMMA

    const int khalf = lane >> 4;
    const int mlane = lane & 15;
    const float* xrow = x + (size_t)(mBase + mlane) * D_IN;

    v8f acc[4] = {v8f{}, v8f{}, v8f{}, v8f{}};

    for (int ks = 0; ks < 8; ++ks) {
        v16h ahi, alo;
#pragma unroll
        for (int p = 0; p < 8; ++p) {
            const int k0 = ks * 32 + kpair_base(p, khalf);
            const float x0 = xrow[k0];
            const float x1 = xrow[k0 + 1];
            const _Float16 h0 = (_Float16)x0, h1 = (_Float16)x1;
            ahi[2 * p]     = h0;
            ahi[2 * p + 1] = h1;
            alo[2 * p]     = (_Float16)(x0 - (float)h0);
            alo[2 * p + 1] = (_Float16)(x1 - (float)h1);
        }
#pragma unroll
        for (int nt = 0; nt < 4; ++nt) {
            const int base = ((ks * 4 + nt) * 32 + lane) * 16;
            const v16h bhi = *reinterpret_cast<const v16h*>(&ldsHi[base]);
            const v16h blo = *reinterpret_cast<const v16h*>(&ldsLo[base]);
            v8f c = acc[nt];
            c = __builtin_amdgcn_wmma_f32_16x16x32_f16(false, ahi, false, bhi,
                                                       (short)0, c, false, false);
            c = __builtin_amdgcn_wmma_f32_16x16x32_f16(false, ahi, false, blo,
                                                       (short)0, c, false, false);
            c = __builtin_amdgcn_wmma_f32_16x16x32_f16(false, alo, false, bhi,
                                                       (short)0, c, false, false);
            acc[nt] = c;
        }
    }

    // Store h tile. C/D layout: lane holds n = 16*nt + (lane&15),
    // VGPR r holds row m = r + 8*(lane>>4).
#pragma unroll
    for (int nt = 0; nt < 4; ++nt) {
#pragma unroll
        for (int r = 0; r < 8; ++r) {
            const int row = mBase + r + 8 * khalf;
            const int col = nt * 16 + mlane;
            hout[(size_t)row * D_OUT + col] = acc[nt][r];
        }
    }

    // Fused epilogue: s1[row] = h[row,:] . a1, s2[row] = h[row,:] . a2
    float a1v[4], a2v[4];
#pragma unroll
    for (int nt = 0; nt < 4; ++nt) {
        a1v[nt] = a[nt * 16 + mlane];
        a2v[nt] = a[D_OUT + nt * 16 + mlane];
    }
#pragma unroll
    for (int r = 0; r < 8; ++r) {
        float p1 = 0.f, p2 = 0.f;
#pragma unroll
        for (int nt = 0; nt < 4; ++nt) {
            p1 += acc[nt][r] * a1v[nt];
            p2 += acc[nt][r] * a2v[nt];
        }
        // reduce across the 16 lanes holding the same row (masks < 16 keep halves)
#pragma unroll
        for (int off = 8; off >= 1; off >>= 1) {
            p1 += __shfl_xor(p1, off, 32);
            p2 += __shfl_xor(p2, off, 32);
        }
        if (mlane == 0) {
            const int row = mBase + r + 8 * khalf;
            s1[row] = p1;
            s2[row] = p2;
        }
    }
}

// --------------------------------------------------------------------------
// Kernel 2: per-edge attention + scatter-add. 64 threads per edge (one per
// output column): h[dst] gather and h'[src] atomic-add are each one coalesced
// 256B line, entirely L2-resident (h table = 25.6 MB << 192 MB L2).
// --------------------------------------------------------------------------
__global__ __launch_bounds__(256)
void gat_edge(const long long* __restrict__ ei, const float* __restrict__ h,
              const float* __restrict__ s1, const float* __restrict__ s2,
              float* __restrict__ hp, float* __restrict__ rs, int nedges)
{
    const long long t = (long long)blockIdx.x * 256 + threadIdx.x;
    const int e = (int)(t >> 6);
    if (e >= nedges) return;
    const int j   = (int)(t & 63);
    const int src = (int)ei[e];
    const int dst = (int)ei[(size_t)nedges + e];

    const float s  = s1[src] + s2[dst];
    const float lr = (s > 0.f) ? s : ALPHA * s;
    const float ev = __expf(-lr);

    unsafeAtomicAdd(&hp[(size_t)src * D_OUT + j], ev * h[(size_t)dst * D_OUT + j]);
    if (j == 0) unsafeAtomicAdd(&rs[src], ev);
}

// --------------------------------------------------------------------------
// Kernel 3: normalize + ELU
// --------------------------------------------------------------------------
__global__ __launch_bounds__(256)
void gat_final(const float* __restrict__ hp, const float* __restrict__ rs,
               float* __restrict__ out, int n)
{
    const int idx = blockIdx.x * 256 + threadIdx.x;
    if (idx >= n * D_OUT) return;
    const float v = hp[idx] / rs[idx >> 6];
    out[idx] = (v > 0.f) ? v : (__expf(v) - 1.0f);
}

extern "C" void kernel_launch(void* const* d_in, const int* in_sizes, int n_in,
                              void* d_out, int out_size, void* d_ws, size_t ws_size,
                              hipStream_t stream) {
    const float*     x  = (const float*)d_in[0];
    const long long* ei = (const long long*)d_in[1];   // int64 (2, E)
    const float*     w  = (const float*)d_in[2];
    const float*     a  = (const float*)d_in[3];

    const int n = in_sizes[0] / D_IN;     // 100000
    const int e = in_sizes[1] / 2;        // 1700000

    // Workspace layout (floats): h[n*64] | s1[n] | s2[n] | rs[n] | hp[n*64]
    float* wsf = (float*)d_ws;
    float* h   = wsf;
    float* s1  = wsf + (size_t)n * D_OUT;
    float* s2  = s1 + n;
    float* rs  = s2 + n;
    float* hp  = rs + n;

    // Zero the accumulators (rs and hp are contiguous) — deterministic per call.
    hipMemsetAsync(rs, 0, sizeof(float) * ((size_t)n * D_OUT + n), stream);

    const int tilesM      = (n + 15) / 16;
    const int gemm_blocks = (tilesM + 7) / 8;
    gat_gemm_wmma<<<gemm_blocks, 256, 0, stream>>>(x, w, a, h, s1, s2, n);

    const long long edge_threads = (long long)e * D_OUT;
    const int edge_blocks = (int)((edge_threads + 255) / 256);
    gat_edge<<<edge_blocks, 256, 0, stream>>>(ei, h, s1, s2, hp, rs, e);

    const int fin_blocks = (n * D_OUT + 255) / 256;
    gat_final<<<fin_blocks, 256, 0, stream>>>(hp, rs, (float*)d_out, n);
}
